// Attention_893353197688
// MI455X (gfx1250) — matrix operand order repaired and checked by reading, not compiled, not run on hardware
//
#include <hip/hip_runtime.h>

typedef __attribute__((ext_vector_type(16))) _Float16 v16h;
typedef __attribute__((ext_vector_type(8)))  float    v8f;

#define BATCH    8
#define CIN      256
#define SEQ      4096           // 64*64
#define DHEAD    128
#define QK_SCALE 0.08838834764831845f   // 128^-0.5
#define SIM_STRIDE 4100         // 4096 + 4 pad dwords: rows r and r+8 land on different LDS banks

// ---------------------------------------------------------------------------
// K0: transpose W[o][c] -> Wt[c][o] (fp32) so the projection reads coalesced.
// ---------------------------------------------------------------------------
__global__ void wt_kernel(const float* __restrict__ W, float* __restrict__ Wt) {
    int c = blockIdx.x;    // 0..255
    int o = threadIdx.x;   // 0..255
    Wt[c * 256 + o] = W[o * 256 + c];
}

// ---------------------------------------------------------------------------
// K1: projection qk[b,o,hw] = sum_c W[o,c] * x[b,c,hw].
// Thread (d, hw) computes q[b,hw,d] (scaled) and k[b,hw,d], fp32 accum,
// stored f16 in [b][n][d] layout (d contiguous -> matches WMMA fragments).
// ---------------------------------------------------------------------------
__global__ __launch_bounds__(256)
void proj_kernel(const float* __restrict__ x, const float* __restrict__ Wt,
                 _Float16* __restrict__ q16, _Float16* __restrict__ k16) {
    int d  = threadIdx.x;                          // 0..127
    int hw = blockIdx.x * 2 + threadIdx.y;         // 0..4095
    int b  = blockIdx.y;
    const float* xb = x + (size_t)b * CIN * SEQ + hw;
    float aq = 0.f, ak = 0.f;
    #pragma unroll 4
    for (int c = 0; c < CIN; ++c) {
        float xv = xb[(size_t)c * SEQ];
        aq = fmaf(Wt[c * 256 + d],       xv, aq);
        ak = fmaf(Wt[c * 256 + 128 + d], xv, ak);
    }
    size_t o = ((size_t)b * SEQ + hw) * DHEAD + d;
    q16[o] = (_Float16)(aq * QK_SCALE);
    k16[o] = (_Float16)ak;
}

// ---------------------------------------------------------------------------
// K2: fused scores + softmax. One WG = (batch, 16 rows of n). 8 waves.
// Each wave: 16 iterations x (2 m-tiles x 4 k-steps) of v_wmma_f32_16x16x32_f16
// with two independent accumulator chains (fills WMMA hazard slots / load
// latency), results staged in a 16 x 4096 fp32 LDS stripe, then row softmax
// in LDS, then one coalesced float4 stream to d_out (the only HBM write of
// the 512MB result).
// ---------------------------------------------------------------------------
__global__ __launch_bounds__(256)
void attn_kernel(const _Float16* __restrict__ q16, const _Float16* __restrict__ k16,
                 float* __restrict__ out) {
    extern __shared__ float smem[];
    float* sim    = smem;                         // 16 * SIM_STRIDE floats
    float* red    = smem + 16 * SIM_STRIDE;       // 16*16 floats
    float* rowinv = red + 256;                    // 16 floats

    const int tid  = threadIdx.x;
    const int lane = tid & 31;
    // scalar wave id -> scalar loop control (no EXEC-mask loop)
    const int wave = __builtin_amdgcn_readfirstlane(tid >> 5);   // 0..7
    const int half = lane >> 4;                   // K-group select
    const int l15  = lane & 15;                   // M (A) / N (B,D) index
    const int n0   = blockIdx.x * 16;             // row tile
    const int b    = blockIdx.y;

    // ---- A fragments: q rows n0..n0+15, all 128 K, contiguous K order, same as B.
    // lane l15 = row; elements 0..15 = K[half*16 .. +16)
    union Frag { v16h v; uint4 u[2]; };
    Frag a[4];
    const _Float16* qb = q16 + ((size_t)(b * SEQ + n0 + l15)) * DHEAD;
    #pragma unroll
    for (int ks = 0; ks < 4; ++ks) {
        a[ks].u[0] = *(const uint4*)(qb + ks * 32 + half * 16);
        a[ks].u[1] = *(const uint4*)(qb + ks * 32 + half * 16 + 8);
    }

    // ---- GEMM: wave w handles m-tile pairs (2w, 2w+1), (2w+16, 2w+17), ...
    // (uniform trip count, EXEC all-ones around every WMMA).
    const _Float16* kbase = k16 + (size_t)b * SEQ * DHEAD + (size_t)l15 * DHEAD + half * 16;
    for (int mt = wave * 2; mt < 256; mt += 16) {
        const int m0 = mt * 16;
        const _Float16* kb0 = kbase + (size_t)m0 * DHEAD;
        const _Float16* kb1 = kb0 + (size_t)16 * DHEAD;
        v8f acc0 = {};
        v8f acc1 = {};
        #pragma unroll
        for (int ks = 0; ks < 4; ++ks) {
            // B fragments: lane l15 = column (= m), 16 consecutive K per half-wave
            Frag b0, b1;
            b0.u[0] = *(const uint4*)(kb0 + ks * 32);
            b0.u[1] = *(const uint4*)(kb0 + ks * 32 + 8);
            b1.u[0] = *(const uint4*)(kb1 + ks * 32);
            b1.u[1] = *(const uint4*)(kb1 + ks * 32 + 8);
            acc0 = __builtin_amdgcn_wmma_f32_16x16x32_f16(
                false, a[ks].v, false, b0.v, (short)0, acc0, false, false);
            acc1 = __builtin_amdgcn_wmma_f32_16x16x32_f16(
                false, a[ks].v, false, b1.v, (short)0, acc1, false, false);
        }
        // D layout: VGPR r -> row (r + 8*half), col l15
        float* dst0 = sim + (size_t)(8 * half) * SIM_STRIDE + m0 + l15;
        #pragma unroll
        for (int r = 0; r < 8; ++r) dst0[r * SIM_STRIDE] = acc0[r];
        float* dst1 = dst0 + 16;
        #pragma unroll
        for (int r = 0; r < 8; ++r) dst1[r * SIM_STRIDE] = acc1[r];
    }
    __syncthreads();

    // ---- softmax over each of the 16 rows; 16 threads per row.
    const int row = tid >> 4;
    const int l16 = tid & 15;
    float* srow = sim + row * SIM_STRIDE;

    float m = -3.402823466e38f;
    for (int j = l16; j < SEQ; j += 16) m = fmaxf(m, srow[j]);
    red[row * 16 + l16] = m;
    __syncthreads();
    float rmax = red[row * 16];
    #pragma unroll
    for (int i = 1; i < 16; ++i) rmax = fmaxf(rmax, red[row * 16 + i]);

    float s = 0.f;
    for (int j = l16; j < SEQ; j += 16) {
        float e = __expf(srow[j] - rmax);
        srow[j] = e;
        s += e;
    }
    __syncthreads();            // all reads of max-partials done before reuse
    red[row * 16 + l16] = s;
    __syncthreads();
    if (l16 == 0) {
        float rs = 0.f;
        #pragma unroll
        for (int i = 0; i < 16; ++i) rs += red[row * 16 + i];
        rowinv[row] = 1.0f / rs;
    }
    __syncthreads();

    // ---- single coalesced writeout: all 256 threads stream each row (float4).
    float* orow = out + ((size_t)(b * SEQ + n0)) * SEQ;
    for (int r = 0; r < 16; ++r) {
        const float  inv = rowinv[r];
        const float* sr  = sim + r * SIM_STRIDE;
        for (int j = tid * 4; j < SEQ; j += 1024) {
            float4 v = *(const float4*)(sr + j);
            v.x *= inv; v.y *= inv; v.z *= inv; v.w *= inv;
            *(float4*)(orow + (size_t)r * SEQ + j) = v;
        }
    }
}

// ---------------------------------------------------------------------------
extern "C" void kernel_launch(void* const* d_in, const int* in_sizes, int n_in,
                              void* d_out, int out_size, void* d_ws, size_t ws_size,
                              hipStream_t stream) {
    const float* x = (const float*)d_in[0];   // (8,256,64,64) fp32
    const float* W = (const float*)d_in[1];   // (256,256) fp32
    float* out = (float*)d_out;               // (8,1,4096,4096) fp32

    char* ws = (char*)d_ws;
    float*    Wt  = (float*)ws;                                   // 256 KB
    _Float16* q16 = (_Float16*)(ws + 262144);                     // 8 MB
    _Float16* k16 = (_Float16*)(ws + 262144 +
                                (size_t)BATCH * SEQ * DHEAD * 2); // 8 MB

    size_t smem = (size_t)(16 * SIM_STRIDE + 256 + 16) * sizeof(float); // ~257 KB
    (void)hipFuncSetAttribute((const void*)attn_kernel,
                              hipFuncAttributeMaxDynamicSharedMemorySize,
                              (int)smem);

    wt_kernel<<<dim3(256), dim3(256), 0, stream>>>(W, Wt);
    proj_kernel<<<dim3(SEQ / 2, BATCH), dim3(128, 2), 0, stream>>>(x, Wt, q16, k16);
    attn_kernel<<<dim3(SEQ / 16, BATCH), dim3(256), smem, stream>>>(q16, k16, out);
}
